// MaskedSelfAttention_26792005992716
// MI455X (gfx1250) — matrix operand order repaired and checked by reading, not compiled, not run on hardware
//
#include <hip/hip_runtime.h>
#include <hip/hip_bf16.h>

// ---------------------------------------------------------------------------
// MaskedSelfAttention for MI455X (gfx1250).
// All matrix math via v_wmma_f32_16x16x32_bf16 (fp32 accumulate).
// GEMMs: block-cooperative B panel staged into LDS with CDNA5 async-to-LDS
// copies (ASYNCcnt) and double buffering; A fragments double-buffered in VGPRs.
// Working set (<100MB) is L2-resident (192MB) -> WMMA-throughput bound.
// ---------------------------------------------------------------------------

#define DEV __device__ __forceinline__

typedef __bf16 bf16_t;
typedef __attribute__((ext_vector_type(16))) __bf16 v16bf;
typedef __attribute__((ext_vector_type(8)))  __bf16 v8bf;
typedef __attribute__((ext_vector_type(8)))  float  v8f;
typedef __attribute__((ext_vector_type(4)))  int    v4i;

typedef __attribute__((address_space(1))) v4i gv4i;   // global 16B chunk
typedef __attribute__((address_space(3))) v4i lv4i;   // LDS 16B chunk

#if defined(__has_builtin)
#if __has_builtin(__builtin_amdgcn_global_load_async_to_lds_b128)
#define HAVE_ASYNC_LDS_BUILTIN 1
#else
#define HAVE_ASYNC_LDS_BUILTIN 0
#endif
#if __has_builtin(__builtin_amdgcn_s_wait_asynccnt)
#define HAVE_WAIT_ASYNC_BUILTIN 1
#else
#define HAVE_WAIT_ASYNC_BUILTIN 0
#endif
#else
#define HAVE_ASYNC_LDS_BUILTIN 0
#define HAVE_WAIT_ASYNC_BUILTIN 0
#endif

constexpr int EMB    = 1024;
constexpr int NH     = 16;
constexpr int HS     = 64;
constexpr int BATCH  = 2;
constexpr int SEQ    = 2048;
constexpr int MROWS  = BATCH * SEQ;   // 4096
constexpr int QKV_N  = 3 * EMB;       // 3072
constexpr int KSTEP  = 32;            // WMMA K depth
constexpr float ATT_SCALE = 0.125f;   // 1/sqrt(64)

DEV int lane_id() { return (int)(threadIdx.x & 31u); }

DEV v8f vzero8() {
  v8f z;
#pragma unroll
  for (int i = 0; i < 8; ++i) z[i] = 0.0f;
  return z;
}

// A fragment (16x32 bf16, MxK): per ISA 7.12.2, lane L holds row M=L%16,
// elems 0..7 -> K=(L/16)*8+i, elems 8..15 -> K=16+(L/16)*8+i.
DEV v16bf load_a_frag(const bf16_t* base, int ld) {
  const int l = lane_id();
  const int m = l & 15, h = l >> 4;
  const bf16_t* p = base + (size_t)m * ld + h * 8;
  v8bf lo = *(const v8bf*)(p);
  v8bf hi = *(const v8bf*)(p + 16);
  return __builtin_shufflevector(lo, hi, 0,1,2,3,4,5,6,7,8,9,10,11,12,13,14,15);
}

// B fragment (32x16 bf16, KxN): lane L holds col N=L%16, elems 0..7 -> K=(L/16)*8+j, elems 8..15 -> K=16+(L/16)*8+j.
// Memory holds B^T (N-major rows, K contiguous, row stride ld).
DEV v16bf load_b_frag(const bf16_t* baseT, int ld) {
  const int l = lane_id();
  const int n = l & 15, h = l >> 4;
  const bf16_t* p = baseT + (size_t)n * ld + h * 8;
  v8bf lo = *(const v8bf*)(p);
  v8bf hi = *(const v8bf*)(p + 16);
  return __builtin_shufflevector(lo, hi, 0,1,2,3,4,5,6,7,8,9,10,11,12,13,14,15);
}

DEV v8f wmma_bf16(v16bf a, v16bf b, v8f c) {
  return __builtin_amdgcn_wmma_f32_16x16x32_bf16(false, a, false, b,
                                                 (short)0, c, false, false);
}

// 16B global -> LDS async copy (ASYNCcnt-tracked, gfx1250)
DEV void async_cp16(const bf16_t* gsrc, bf16_t* ldst) {
#if HAVE_ASYNC_LDS_BUILTIN
  __builtin_amdgcn_global_load_async_to_lds_b128(
      (gv4i*)(void*)gsrc, (lv4i*)ldst, 0, 0);
#else
  unsigned loff = (unsigned)(unsigned long long)(lv4i*)ldst;
  asm volatile("global_load_async_to_lds_b128 %0, %1, off"
               :: "v"(loff), "v"(gsrc) : "memory");
#endif
}

DEV void wait_asynccnt0() {
#if HAVE_WAIT_ASYNC_BUILTIN
  __builtin_amdgcn_s_wait_asynccnt(0);
#else
  asm volatile("s_wait_asynccnt 0" ::: "memory");
#endif
}

// ---------------------------------------------------------------------------
// Prologue kernels: precision conversion + layout changes
// ---------------------------------------------------------------------------

__global__ void cvt_bf16(const float* __restrict__ X, bf16_t* __restrict__ Y, int n) {
  int i = blockIdx.x * blockDim.x + threadIdx.x;
  if (i < n) Y[i] = (bf16_t)X[i];
}

// W[K][N] fp32 -> Wt[N][K] bf16 (GEMM B-fragments become K-contiguous)
__global__ void cvt_transpose(const float* __restrict__ W, bf16_t* __restrict__ Wt,
                              int K, int N) {
  int i = blockIdx.x * blockDim.x + threadIdx.x;
  if (i >= K * N) return;
  int k = i / N, n = i - k * N;
  Wt[(size_t)n * K + k] = (bf16_t)W[i];
}

// qkv[b,t,3C] (V slice) -> Vt[b,h,d,t] so P@V B-fragments are K(=t)-contiguous
__global__ void build_vt(const bf16_t* __restrict__ qkv, bf16_t* __restrict__ vt) {
  int i = blockIdx.x * blockDim.x + threadIdx.x;
  if (i >= BATCH * SEQ * EMB) return;
  int d = i & (HS - 1);
  int h = (i >> 6) & (NH - 1);
  int t = (i / EMB) % SEQ;
  int b = i / (SEQ * EMB);
  bf16_t v = qkv[(size_t)(b * SEQ + t) * QKV_N + 2 * EMB + h * HS + d];
  vt[(size_t)((b * NH + h) * HS + d) * SEQ + t] = v;
}

// ---------------------------------------------------------------------------
// GEMM: C[M,N] = A[M,K] @ B[K,N].  Block = 4 waves -> 128x64 tile.
// Per k-step (K=32): B panel (64 N-rows x 32 K) async-copied to LDS, double
// buffered; each wave holds 2 A fragments double-buffered in VGPRs and runs
// an 8-WMMA burst against 4 LDS-resident B fragments (all ds_loads issued
// before the burst so one s_wait_dscnt covers them).
// ---------------------------------------------------------------------------

__global__ void __launch_bounds__(128)
gemm_bf16_wmma(const bf16_t* __restrict__ A, const bf16_t* __restrict__ Bt,
               bf16_t* __restrict__ Cb, float* __restrict__ Cf,
               const float* __restrict__ bias, int Ndim, int K) {
  __shared__ __align__(16) bf16_t ldsB[2][64 * KSTEP];   // 2 x 4KB

  const int nb = Ndim >> 6;
  const int bm = (int)(blockIdx.x / nb) * 128;
  const int bn = (int)(blockIdx.x % nb) * 64;
  const int w  = (int)(threadIdx.x >> 5);
  const int tm = bm + w * 32;

  const int l    = lane_id();
  const int half = l >> 4;
  const int n0   = l & 15;
  const int tid  = (int)threadIdx.x;

  v8f acc[2][4];
#pragma unroll
  for (int i = 0; i < 2; ++i)
#pragma unroll
    for (int t = 0; t < 4; ++t) acc[i][t] = vzero8();

  const bf16_t* aA = A + (size_t)tm * K;

  // cooperative async fill of one 64x32 B^T panel chunk into LDS
  auto fill_b = [&](int kc, bf16_t* dst) {
#pragma unroll
    for (int i = 0; i < 2; ++i) {
      int c = tid + i * 128;            // 256 chunks of 16B
      int n = c >> 2, s = (c & 3) * 8;  // row n, 8-elem sub-chunk s
      async_cp16(Bt + (size_t)(bn + n) * K + kc + s, dst + n * KSTEP + s);
    }
  };

  // prologue: stage k-chunk 0, preload A fragments for k-chunk 0
  fill_b(0, &ldsB[0][0]);
  v16bf a0 = load_a_frag(aA, K);
  v16bf a1 = load_a_frag(aA + (size_t)16 * K, K);
  wait_asynccnt0();
  __syncthreads();

  const int nk = K / KSTEP;
  for (int ks = 0; ks < nk; ++ks) {
    const int  buf  = ks & 1;
    const bool more = (ks + 1) < nk;

    // issue next B panel (overlaps with this step's WMMAs)
    if (more) fill_b((ks + 1) * KSTEP, &ldsB[buf ^ 1][0]);

    // prefetch next A fragments into registers
    v16bf an0 = a0, an1 = a1;
    if (more) {
      an0 = load_a_frag(aA + (ks + 1) * KSTEP, K);
      an1 = load_a_frag(aA + (size_t)16 * K + (ks + 1) * KSTEP, K);
    }

    // issue all 8 ds_load_b128 first, then the 8-WMMA burst
    v16bf bf[4];
#pragma unroll
    for (int t = 0; t < 4; ++t)
      bf[t] = load_b_frag(&ldsB[buf][t * 16 * KSTEP], KSTEP);
#pragma unroll
    for (int t = 0; t < 4; ++t) {
      acc[0][t] = wmma_bf16(a0, bf[t], acc[0][t]);
      acc[1][t] = wmma_bf16(a1, bf[t], acc[1][t]);
    }
    a0 = an0; a1 = an1;

    wait_asynccnt0();   // next panel landed (this wave)
    __syncthreads();    // all waves: panel complete + done reading old buffer
  }

  // C/D layout: elem r -> row = half*8 + r, col = lane&15 (ISA 7.12.2)
#pragma unroll
  for (int i = 0; i < 2; ++i)
#pragma unroll
    for (int t = 0; t < 4; ++t)
#pragma unroll
      for (int r = 0; r < 8; ++r) {
        int gm = tm + i * 16 + half * 8 + r;
        int gn = bn + t * 16 + n0;
        float v = acc[i][t][r];
        if (Cb) Cb[(size_t)gm * Ndim + gn] = (bf16_t)v;
        else    Cf[(size_t)gm * Ndim + gn] = v + bias[gn];
      }
}

// ---------------------------------------------------------------------------
// Causal flash attention. Block = 4 waves; wave w owns 16 Q rows of one (b,h).
// KV processed 32 keys/iteration: S = QK^T (4 WMMAs), online softmax with
// 16-lane shuffle-group row stats, P transposed C->A layout through per-wave
// LDS, O += P@V (4 WMMAs). Fragment loads are issued ahead of each WMMA burst.
// ---------------------------------------------------------------------------

__global__ void __launch_bounds__(128)
attn_causal_wmma(const bf16_t* __restrict__ qkv, const bf16_t* __restrict__ vt,
                 bf16_t* __restrict__ out) {
  __shared__ __align__(16) bf16_t pbuf[4][16][32];

  const int w  = (int)(threadIdx.x >> 5);
  const int qb = blockIdx.x & 31;           // SEQ/64 = 32 Q-blocks
  const int bh = blockIdx.x >> 5;
  const int h  = bh & (NH - 1);
  const int b  = bh >> 4;
  const int q0 = (qb * 4 + w) * 16;

  const int l    = lane_id();
  const int half = l >> 4;
  const int nn   = l & 15;

  float rmax[8], rsum[8];
  v8f o[4];
#pragma unroll
  for (int r = 0; r < 8; ++r) { rmax[r] = -1e30f; rsum[r] = 0.0f; }
#pragma unroll
  for (int t = 0; t < 4; ++t) o[t] = vzero8();

  const bf16_t* qbase = qkv + (size_t)(b * SEQ + q0) * QKV_N + h * HS;
  v16bf qa0 = load_a_frag(qbase,      QKV_N);
  v16bf qa1 = load_a_frag(qbase + 32, QKV_N);

  const bf16_t* vhead = vt + (size_t)((b * NH + h) * HS) * SEQ;

  for (int kv0 = 0; kv0 <= q0 + 15; kv0 += 32) {
    const bf16_t* kbase = qkv + (size_t)(b * SEQ + kv0) * QKV_N + EMB + h * HS;

    // issue all K-fragment loads, then the 4-WMMA burst
    v16bf kb[4];
    kb[0] = load_b_frag(kbase,                           QKV_N);
    kb[1] = load_b_frag(kbase + 32,                      QKV_N);
    kb[2] = load_b_frag(kbase + (size_t)16 * QKV_N,      QKV_N);
    kb[3] = load_b_frag(kbase + (size_t)16 * QKV_N + 32, QKV_N);

    v8f s0 = vzero8(), s1 = vzero8();
    s0 = wmma_bf16(qa0, kb[0], s0);
    s0 = wmma_bf16(qa1, kb[1], s0);
    s1 = wmma_bf16(qa0, kb[2], s1);
    s1 = wmma_bf16(qa1, kb[3], s1);

#pragma unroll
    for (int r = 0; r < 8; ++r) {
      int qrow = q0 + half * 8 + r;
      int c0 = kv0 + nn;
      s0[r] = (c0      <= qrow) ? s0[r] * ATT_SCALE : -1e30f;
      s1[r] = (c0 + 16 <= qrow) ? s1[r] * ATT_SCALE : -1e30f;
    }

#pragma unroll
    for (int r = 0; r < 8; ++r) {
      float tmax = fmaxf(s0[r], s1[r]);
      tmax = fmaxf(tmax, __shfl_xor(tmax, 1, 32));
      tmax = fmaxf(tmax, __shfl_xor(tmax, 2, 32));
      tmax = fmaxf(tmax, __shfl_xor(tmax, 4, 32));
      tmax = fmaxf(tmax, __shfl_xor(tmax, 8, 32));
      float nm    = fmaxf(rmax[r], tmax);
      float alpha = __expf(rmax[r] - nm);
      rmax[r] = nm;
      float p0 = __expf(s0[r] - nm);
      float p1 = __expf(s1[r] - nm);
      s0[r] = p0; s1[r] = p1;
      float ps = p0 + p1;
      ps += __shfl_xor(ps, 1, 32);
      ps += __shfl_xor(ps, 2, 32);
      ps += __shfl_xor(ps, 4, 32);
      ps += __shfl_xor(ps, 8, 32);
      rsum[r] = rsum[r] * alpha + ps;
      o[0][r] *= alpha; o[1][r] *= alpha; o[2][r] *= alpha; o[3][r] *= alpha;
    }

    // P (C-layout) -> per-wave LDS [16][32] -> A fragment
#pragma unroll
    for (int r = 0; r < 8; ++r) {
      int m = half * 8 + r;
      pbuf[w][m][nn]      = (bf16_t)s0[r];
      pbuf[w][m][nn + 16] = (bf16_t)s1[r];
    }
    __builtin_amdgcn_wave_barrier();
    asm volatile("s_wait_dscnt 0" ::: "memory");   // DS in-order per wave

    v16bf pa = load_a_frag(&pbuf[w][0][0], 32);

    // issue all V-fragment loads, then the 4-WMMA burst
    const bf16_t* vbase = vhead + kv0;
    v16bf vb[4];
#pragma unroll
    for (int dt = 0; dt < 4; ++dt)
      vb[dt] = load_b_frag(vbase + (size_t)(dt * 16) * SEQ, SEQ);
#pragma unroll
    for (int dt = 0; dt < 4; ++dt)
      o[dt] = wmma_bf16(pa, vb[dt], o[dt]);

    __builtin_amdgcn_wave_barrier();
  }

#pragma unroll
  for (int dt = 0; dt < 4; ++dt)
#pragma unroll
    for (int r = 0; r < 8; ++r) {
      int m    = half * 8 + r;
      int trow = q0 + m;
      int c    = h * HS + dt * 16 + nn;
      out[(size_t)(b * SEQ + trow) * EMB + c] = (bf16_t)(o[dt][r] / rsum[r]);
    }
}

// ---------------------------------------------------------------------------

extern "C" void kernel_launch(void* const* d_in, const int* in_sizes, int n_in,
                              void* d_out, int out_size, void* d_ws, size_t ws_size,
                              hipStream_t stream) {
  (void)in_sizes; (void)n_in; (void)out_size; (void)ws_size;

  const float* x     = (const float*)d_in[0];  // [B,T,C]
  const float* Wqkv  = (const float*)d_in[1];  // [C,3C]
  const float* Wproj = (const float*)d_in[2];  // [C,C]
  const float* bproj = (const float*)d_in[3];  // [C]
  float* out = (float*)d_out;                  // [B,T,C] fp32

  char*  ws  = (char*)d_ws;
  size_t off = 0;
  auto carve = [&](size_t bytes) {
    void* p = ws + off;
    off = (off + bytes + 255) & ~(size_t)255;
    return p;
  };
  bf16_t* xb     = (bf16_t*)carve((size_t)MROWS * EMB   * 2);  //  8 MB
  bf16_t* wqkvT  = (bf16_t*)carve((size_t)QKV_N * EMB   * 2);  //  6 MB
  bf16_t* wprojT = (bf16_t*)carve((size_t)EMB   * EMB   * 2);  //  2 MB
  bf16_t* qkvb   = (bf16_t*)carve((size_t)MROWS * QKV_N * 2);  // 24 MB
  bf16_t* vtb    = (bf16_t*)carve((size_t)BATCH * NH * HS * SEQ * 2); // 8 MB
  bf16_t* attnb  = (bf16_t*)carve((size_t)MROWS * EMB   * 2);  //  8 MB

  const int TPB = 256;

  cvt_bf16<<<(MROWS * EMB + TPB - 1) / TPB, TPB, 0, stream>>>(x, xb, MROWS * EMB);
  cvt_transpose<<<(EMB * QKV_N + TPB - 1) / TPB, TPB, 0, stream>>>(Wqkv, wqkvT, EMB, QKV_N);
  cvt_transpose<<<(EMB * EMB   + TPB - 1) / TPB, TPB, 0, stream>>>(Wproj, wprojT, EMB, EMB);

  // QKV projection: [4096,1024] @ [1024,3072] -> bf16
  gemm_bf16_wmma<<<(MROWS / 128) * (QKV_N / 64), 128, 0, stream>>>(
      xb, wqkvT, qkvb, nullptr, nullptr, QKV_N, EMB);

  build_vt<<<(BATCH * SEQ * EMB + TPB - 1) / TPB, TPB, 0, stream>>>(qkvb, vtb);

  attn_causal_wmma<<<BATCH * NH * (SEQ / 64), 128, 0, stream>>>(qkvb, vtb, attnb);

  // output projection + bias -> fp32 d_out
  gemm_bf16_wmma<<<(MROWS / 128) * (EMB / 64), 128, 0, stream>>>(
      attnb, wprojT, nullptr, out, bproj, EMB, EMB);
}
